// TFMambaBlock_29463475651161
// MI455X (gfx1250) — compile-verified, matrix operand order checked by reading, not run on hardware
//
#include <hip/hip_runtime.h>
#include <hip/hip_bf16.h>

// ---------------------------------------------------------------------------
// TF-Mamba block for MI455X (gfx1250, wave32, WMMA).
//   Dims: B=2, C(d_model)=64, T=256, F=128, D_INNER=256, D_STATE=16,
//         DT_RANK=4, D_CONV=4.  Tokens per stage N = 65536.
// ---------------------------------------------------------------------------

typedef __bf16  bf16_t;
typedef __attribute__((ext_vector_type(16))) __bf16 v16bf;
typedef __attribute__((ext_vector_type(8)))  float  v8f;

#define NTOK   65536
#define DM     64
#define DIN    256
#define DSTATE 16

// ---------------------------------------------------------------------------
// Fragment loader: reads 16 f32 values (row-major, row `row`, cols k0..)
// and packs them as a CDNA5 16-bit A/B-matrix fragment (bf16).
// Layout (ISA 7.12.2, 16-bit A 16x32):
//   lanes 0-15  : elements 0-7 = K k0+0..7,  elements 8-15 = K k0+16..23
//   lanes 16-31 : elements 0-7 = K k0+8..15, elements 8-15 = K k0+24..31
// B (32x16, column n = lane&15) uses the identical packing over W rows.
// ---------------------------------------------------------------------------
__device__ __forceinline__ v16bf load_frag_bf16(const float* __restrict__ base,
                                                int row, int ld, int k0, int lane) {
  const float* p = base + row * ld + k0 + ((lane & 16) ? 8 : 0);
  v16bf f;
#pragma unroll
  for (int j = 0; j < 8; ++j) f[j] = (__bf16)p[j];
#pragma unroll
  for (int j = 0; j < 8; ++j) f[8 + j] = (__bf16)p[16 + j];
  return f;
}

// ---------------------------------------------------------------------------
// Generic WMMA GEMM:  C[m, col0+n] = sum_k A[m,k] * W[n,k]  (+bias)(+resid)
// A: (M x K) f32 row-major, W: (N x K) f32 row-major (weights as stored).
// One wave computes a 16 x (16*TN) tile.  grid = (M/16, N/(16*TN)), block=32.
// ---------------------------------------------------------------------------
template <int TN>
__global__ void wmma_gemm_kernel(const float* __restrict__ A, int lda,
                                 const float* __restrict__ W, int ldw,
                                 float* __restrict__ C, int ldc, int col0,
                                 const float* __restrict__ bias,
                                 const float* __restrict__ resid, int ldr,
                                 int K) {
  const int lane = threadIdx.x;
  const int mt   = blockIdx.x;
  const int nt0  = blockIdx.y * TN;
  const int arow = mt * 16 + (lane & 15);

  v8f acc[TN];
#pragma unroll
  for (int i = 0; i < TN; ++i) acc[i] = (v8f){0, 0, 0, 0, 0, 0, 0, 0};

  for (int k0 = 0; k0 < K; k0 += 32) {
    v16bf a = load_frag_bf16(A, arow, lda, k0, lane);
#pragma unroll
    for (int i = 0; i < TN; ++i) {
      v16bf b = load_frag_bf16(W, (nt0 + i) * 16 + (lane & 15), ldw, k0, lane);
      acc[i] = __builtin_amdgcn_wmma_f32_16x16x32_bf16(
          /*neg_a=*/false, a, /*neg_b=*/false, b,
          /*c_mod=*/(short)0, acc[i], /*reuse_a=*/false, /*reuse_b=*/false);
    }
  }

  // C/D layout: lanes 0-15 -> M = r,   N = lane
  //             lanes16-31 -> M = 8+r, N = lane-16
  const int mb = mt * 16 + ((lane & 16) ? 8 : 0);
  const int n  = lane & 15;
#pragma unroll
  for (int i = 0; i < TN; ++i) {
    const int col = (nt0 + i) * 16 + n;
    const float bi = bias ? bias[col] : 0.0f;
#pragma unroll
    for (int r = 0; r < 8; ++r) {
      const int m = mb + r;
      float v = acc[i][r] + bi;
      if (resid) v += resid[m * ldr + col];
      C[m * ldc + col0 + col] = v;
    }
  }
}

// ---------------------------------------------------------------------------
// Depthwise causal conv (k=4) + bias + SiLU.  Reads x_in = xz[:, 0:256].
// dir=0: taps t-3..t with w[0..3];  dir=1 (reversed seq): taps t..t+3
// with w[3..0] (equivalent to running the conv on the flipped sequence).
// ---------------------------------------------------------------------------
__global__ void conv_silu_kernel(const float* __restrict__ xz,
                                 const float* __restrict__ conv_w,
                                 const float* __restrict__ conv_b,
                                 float* __restrict__ xc, int L, int dir) {
  const int idx = blockIdx.x * blockDim.x + threadIdx.x;  // over NTOK*256
  const int d   = idx & 255;
  const int tok = idx >> 8;
  const int t   = tok % L;
  const int s   = tok / L;
  const float* w = conv_w + d * 4;
  float acc = conv_b[d];
  if (dir == 0) {
#pragma unroll
    for (int k = 0; k < 4; ++k) {
      const int tt = t - 3 + k;
      if (tt >= 0) acc += w[k] * xz[(s * L + tt) * 512 + d];
    }
  } else {
#pragma unroll
    for (int k = 0; k < 4; ++k) {
      const int tt = t + 3 - k;
      if (tt < L) acc += w[k] * xz[(s * L + tt) * 512 + d];
    }
  }
  xc[tok * 256 + d] = acc / (1.0f + __expf(-acc));  // SiLU
}

// Zero-pad x_proj_w (36 x 256) into (48 x 256) so the GEMM is 3 full N-tiles.
__global__ void pad_xproj_kernel(const float* __restrict__ w,
                                 float* __restrict__ wp) {
  const int idx = blockIdx.x * blockDim.x + threadIdx.x;  // 48*256
  const int r   = idx >> 8;
  wp[idx] = (r < 36) ? w[idx] : 0.0f;
}

// ---------------------------------------------------------------------------
// Selective scan, fused with dt_proj+softplus, +x*D, *silu(z).
// One block = one sequence, 256 threads = 256 channels, 16 f32 states each.
// xc is read (conv output) and overwritten in place with the gated y.
// Per-token dt_raw(4)/B(16)/C(16) broadcast via LDS.
// ---------------------------------------------------------------------------
__global__ void scan_kernel(const float* __restrict__ xz,
                            float* __restrict__ xc,
                            const float* __restrict__ xdbl,
                            const float* __restrict__ dt_w,
                            const float* __restrict__ dt_b,
                            const float* __restrict__ A_log,
                            const float* __restrict__ Dp,
                            int L, int dir) {
  __shared__ float sh[36];
  const int d = threadIdx.x;
  const int s = blockIdx.x;

  float A[DSTATE];
#pragma unroll
  for (int n = 0; n < DSTATE; ++n) A[n] = -__expf(A_log[d * DSTATE + n]);
  const float w0 = dt_w[d * 4 + 0], w1 = dt_w[d * 4 + 1];
  const float w2 = dt_w[d * 4 + 2], w3 = dt_w[d * 4 + 3];
  const float bd = dt_b[d];
  const float Dd = Dp[d];

  float h[DSTATE];
#pragma unroll
  for (int n = 0; n < DSTATE; ++n) h[n] = 0.0f;

  for (int i = 0; i < L; ++i) {
    const int t   = dir ? (L - 1 - i) : i;
    const int tok = s * L + t;
    __syncthreads();
    if (d < 36) sh[d] = xdbl[tok * 48 + d];
    __syncthreads();

    float dt = w0 * sh[0] + w1 * sh[1] + w2 * sh[2] + w3 * sh[3] + bd;
    dt = (dt > 20.0f) ? dt : log1pf(__expf(dt));  // softplus
    const float xv  = xc[tok * 256 + d];
    const float dtx = dt * xv;
    float y = 0.0f;
#pragma unroll
    for (int n = 0; n < DSTATE; ++n) {
      h[n] = __expf(dt * A[n]) * h[n] + dtx * sh[4 + n];
      y += h[n] * sh[20 + n];
    }
    y += xv * Dd;
    const float z = xz[tok * 512 + 256 + d];
    y *= z / (1.0f + __expf(-z));  // * silu(z)
    xc[tok * 256 + d] = y;
  }
}

// ---------------------------------------------------------------------------
// LayerNorm over 64 features (wave32: 2 features/lane, shuffle reduce)
// + stage residual.
// ---------------------------------------------------------------------------
__global__ void ln_kernel(const float* __restrict__ pre,
                          const float* __restrict__ g,
                          const float* __restrict__ b,
                          const float* __restrict__ resid,
                          float* __restrict__ out) {
  const int token = blockIdx.x * 8 + (threadIdx.x >> 5);
  const int lane  = threadIdx.x & 31;
  const int base  = token * 64;
  float v0 = pre[base + lane], v1 = pre[base + lane + 32];
  float s = v0 + v1, q = v0 * v0 + v1 * v1;
#pragma unroll
  for (int o = 16; o > 0; o >>= 1) {
    s += __shfl_xor(s, o, 32);
    q += __shfl_xor(q, o, 32);
  }
  const float mean = s * (1.0f / 64.0f);
  const float var  = q * (1.0f / 64.0f) - mean * mean;
  const float inv  = rsqrtf(var + 1e-5f);
  out[base + lane]      = (v0 - mean) * inv * g[lane]      + b[lane]      + resid[base + lane];
  out[base + lane + 32] = (v1 - mean) * inv * g[lane + 32] + b[lane + 32] + resid[base + lane + 32];
}

// ---------------------------------------------------------------------------
// Layout shuffles.  x: (B=2, C=64, T=256, F=128).
// Xtime token = (b*128+f)*256+t ; Xfreq token = (b*256+t)*128+f.
// ---------------------------------------------------------------------------
__global__ void t0_kernel(const float* __restrict__ x, float* __restrict__ Xt) {
  const int idx = blockIdx.x * blockDim.x + threadIdx.x;  // ((b*F+f)*T+t)*64+c
  const int c = idx & 63;
  const int t = (idx >> 6) & 255;
  const int f = (idx >> 14) & 127;
  const int b = idx >> 21;
  Xt[idx] = x[((b * 64 + c) * 256 + t) * 128 + f];
}

__global__ void t1_kernel(const float* __restrict__ Yt, float* __restrict__ Xf) {
  const int idx = blockIdx.x * blockDim.x + threadIdx.x;  // ((b*T+t)*F+f)*64+c
  const int c = idx & 63;
  const int f = (idx >> 6) & 127;
  const int t = (idx >> 13) & 255;
  const int b = idx >> 21;
  Xf[idx] = Yt[((b * 128 + f) * 256 + t) * 64 + c];
}

__global__ void t2_kernel(const float* __restrict__ Yf, float* __restrict__ out) {
  const int idx = blockIdx.x * blockDim.x + threadIdx.x;  // ((b*C+c)*T+t)*F+f
  const int f = idx & 127;
  const int t = (idx >> 7) & 255;
  const int c = (idx >> 15) & 63;
  const int b = idx >> 21;
  out[idx] = Yf[((b * 256 + t) * 128 + f) * 64 + c];
}

// ---------------------------------------------------------------------------
// Driver.  Param order (setup_inputs insertion order):
//   d_in[0] = x
//   per block (time @1, freq @23):
//     fw mamba (9): in_proj_w, conv_w, conv_b, x_proj_w, dt_proj_w,
//                   dt_proj_b, A_log, D, out_proj_w
//     bw mamba (9), proj_w, proj_b, ln_g, ln_b
// ---------------------------------------------------------------------------
extern "C" void kernel_launch(void* const* d_in, const int* in_sizes, int n_in,
                              void* d_out, int out_size, void* d_ws, size_t ws_size,
                              hipStream_t stream) {
  auto F32 = [&](int i) { return (const float*)d_in[i]; };
  const float* x = F32(0);

  float* ws   = (float*)d_ws;
  float* Xbuf = ws;                    // NTOK*64   stage input (residual source)
  float* Ybuf = Xbuf + NTOK * 64;      // NTOK*64   stage output
  float* xz   = Ybuf + NTOK * 64;      // NTOK*512  in_proj output (x_in | z)
  float* xc   = xz + NTOK * 512;       // NTOK*256  conv out, then scan out (in place)
  float* xdbl = xc + NTOK * 256;       // NTOK*48   x_proj output (dt|B|C, padded)
  float* ocat = xdbl + NTOK * 48;      // NTOK*128  [out_fw | out_bw]
  float* pre  = ocat + NTOK * 128;     // NTOK*64   pre-layernorm
  float* wpad = pre + NTOK * 64;       // 48*256    padded x_proj_w

  auto run_mamba_dir = [&](int pb, int S, int L, int dir) {
    const float* in_proj = F32(pb + 0);
    const float* conv_w  = F32(pb + 1);
    const float* conv_b  = F32(pb + 2);
    const float* x_proj  = F32(pb + 3);
    const float* dt_w    = F32(pb + 4);
    const float* dt_b    = F32(pb + 5);
    const float* A_log   = F32(pb + 6);
    const float* Dp      = F32(pb + 7);
    const float* out_prj = F32(pb + 8);

    // in_proj: (NTOK x 64) @ (512 x 64)^T -> xz (NTOK x 512)
    wmma_gemm_kernel<4><<<dim3(NTOK / 16, 512 / 64), 32, 0, stream>>>(
        Xbuf, 64, in_proj, 64, xz, 512, 0, nullptr, nullptr, 0, 64);
    // depthwise conv + SiLU -> xc
    conv_silu_kernel<<<(NTOK * 256) / 256, 256, 0, stream>>>(
        xz, conv_w, conv_b, xc, L, dir);
    // x_proj (padded 36->48): (NTOK x 256) @ (48 x 256)^T -> xdbl
    pad_xproj_kernel<<<(48 * 256) / 256, 256, 0, stream>>>(x_proj, wpad);
    wmma_gemm_kernel<3><<<dim3(NTOK / 16, 1), 32, 0, stream>>>(
        xc, 256, wpad, 256, xdbl, 48, 0, nullptr, nullptr, 0, 256);
    // selective scan (fused dt_proj/softplus/D/silu(z)), in place on xc
    scan_kernel<<<S, 256, 0, stream>>>(xz, xc, xdbl, dt_w, dt_b, A_log, Dp, L, dir);
    // out_proj: (NTOK x 256) @ (64 x 256)^T + x  -> ocat[:, dir*64 ..]
    wmma_gemm_kernel<4><<<dim3(NTOK / 16, 1), 32, 0, stream>>>(
        xc, 256, out_prj, 256, ocat, 128, dir * 64, nullptr, Xbuf, 64, 256);
  };

  auto run_block = [&](int pb, int S, int L) {
    run_mamba_dir(pb + 0, S, L, 0);   // forward
    run_mamba_dir(pb + 9, S, L, 1);   // backward (reversed indexing)
    const float* proj_w = F32(pb + 18);
    const float* proj_b = F32(pb + 19);
    const float* ln_g   = F32(pb + 20);
    const float* ln_b   = F32(pb + 21);
    // block proj: (NTOK x 128) @ (64 x 128)^T + b -> pre
    wmma_gemm_kernel<4><<<dim3(NTOK / 16, 1), 32, 0, stream>>>(
        ocat, 128, proj_w, 128, pre, 64, 0, proj_b, nullptr, 0, 128);
    // layernorm + stage residual -> Ybuf
    ln_kernel<<<NTOK / 8, 256, 0, stream>>>(pre, ln_g, ln_b, Xbuf, Ybuf);
  };

  // Stage 1: time axis (S = B*F = 256 sequences of L = 256)
  t0_kernel<<<(NTOK * 64) / 256, 256, 0, stream>>>(x, Xbuf);
  run_block(1, 256, 256);

  // Stage 2: freq axis (S = B*T = 512 sequences of L = 128)
  t1_kernel<<<(NTOK * 64) / 256, 256, 0, stream>>>(Ybuf, Xbuf);
  run_block(23, 512, 128);

  // Back to (B, C, T, F)
  t2_kernel<<<(NTOK * 64) / 256, 256, 0, stream>>>(Ybuf, (float*)d_out);
}
